// GIN_v2_38792144617976
// MI455X (gfx1250) — matrix operand order repaired and checked by reading, not compiled, not run on hardware
//
#include <hip/hip_runtime.h>

#define NN 50000
#define NE 600000
#define DD 128
#define HH 256
#define NEG 0.2f

typedef __attribute__((ext_vector_type(16))) _Float16 v16h;
typedef __attribute__((ext_vector_type(8)))  _Float16 v8h;
typedef __attribute__((ext_vector_type(8)))  float    v8f;

// ---------------- zero agg (layer 0 only; later layers fuse zeroing into MLP) ----
__global__ __launch_bounds__(256) void gin_zero_kernel(float* __restrict__ p, int n4) {
  int i = blockIdx.x * 256 + threadIdx.x;
  if (i < n4) ((float4*)p)[i] = make_float4(0.f, 0.f, 0.f, 0.f);
}

// ---------------- scatter-add: agg[dst] += h[src] ----------------
// one wave per edge, each lane moves 4 contiguous floats (coalesced 512B row gather)
__global__ __launch_bounds__(256) void gin_scatter_kernel(
    const float* __restrict__ h, const long long* __restrict__ ei,
    float* __restrict__ agg) {
  int t = blockIdx.x * 256 + threadIdx.x;
  int e = t >> 5;
  if (e >= NE) return;
  int lane = t & 31;
  int s = (int)ei[e];
  int d = (int)ei[(size_t)NE + e];
  float4 v = *(const float4*)(h + (size_t)s * DD + lane * 4);
  float* a = agg + (size_t)d * DD + lane * 4;
  (void)__hip_atomic_fetch_add(a + 0, v.x, __ATOMIC_RELAXED, __HIP_MEMORY_SCOPE_AGENT);
  (void)__hip_atomic_fetch_add(a + 1, v.y, __ATOMIC_RELAXED, __HIP_MEMORY_SCOPE_AGENT);
  (void)__hip_atomic_fetch_add(a + 2, v.z, __ATOMIC_RELAXED, __HIP_MEMORY_SCOPE_AGENT);
  (void)__hip_atomic_fetch_add(a + 3, v.w, __ATOMIC_RELAXED, __HIP_MEMORY_SCOPE_AGENT);
}

// branchless LeakyReLU: for 0 < slope < 1, lrelu(x) == max(x, slope*x)
__device__ __forceinline__ float gin_lrelu(float x) { return fmaxf(x, NEG * x); }

// pack 8 floats of (h+agg) into 8 f16 halves (one A-fragment chunk)
__device__ __forceinline__ v8h gin_cvt8(const float* hp, const float* gp) {
  float4 x0 = *(const float4*)hp, x1 = *(const float4*)(hp + 4);
  float4 y0 = *(const float4*)gp, y1 = *(const float4*)(gp + 4);
  v8h r;
  r[0] = (_Float16)(x0.x + y0.x); r[1] = (_Float16)(x0.y + y0.y);
  r[2] = (_Float16)(x0.z + y0.z); r[3] = (_Float16)(x0.w + y0.w);
  r[4] = (_Float16)(x1.x + y1.x); r[5] = (_Float16)(x1.y + y1.y);
  r[6] = (_Float16)(x1.z + y1.z); r[7] = (_Float16)(x1.w + y1.w);
  return r;
}

__device__ __forceinline__ v16h gin_join(v8h lo, v8h hi) {
  v16h a;
#pragma unroll
  for (int j = 0; j < 8; ++j) { a[j] = lo[j]; a[8 + j] = hi[j]; }
  return a;
}

// ---------------- fused GIN MLP: out = act((lrelu((h+agg)W1+b1))W2+b2) ----------------
// also re-zeroes this wave's agg rows for the next layer's scatter
__global__ __launch_bounds__(256) void gin_mlp_kernel(
    const float* __restrict__ h, float* __restrict__ agg,
    float* __restrict__ out,
    const float* __restrict__ W1g, const float* __restrict__ b1g,
    const float* __restrict__ W2g, const float* __restrict__ b2g,
    int outer_relu) {
  extern __shared__ __align__(32) char smem[];
  _Float16* sW1t = (_Float16*)smem;      // [HH][DD]  W1 transposed, f16 (64 KB)
  _Float16* sW2t = sW1t + HH * DD;       // [DD][HH]  W2 transposed, f16 (64 KB)
  _Float16* sHid = sW2t + DD * HH;       // [128][HH] per-wave hidden tiles (64 KB)

  const int tid = threadIdx.x;

  // stage weights transposed: sW[o][k] = W[k][o], converted to f16
  for (int idx = tid; idx < HH * DD; idx += 256) {
    int o = idx >> 7, i = idx & (DD - 1);
    sW1t[idx] = (_Float16)W1g[i * HH + o];
  }
  for (int idx = tid; idx < DD * HH; idx += 256) {
    int o = idx >> 8, i = idx & (HH - 1);
    sW2t[idx] = (_Float16)W2g[i * DD + o];
  }
  __syncthreads();

  const int wave = tid >> 5;
  const int lane = tid & 31;
  const int l15  = lane & 15;
  const int hihalf = lane >> 4;     // 0: lanes 0-15, 1: lanes 16-31
  const int off  = hihalf * 8;      // A-fragment K sub-offset per ISA layout
  const int m0   = blockIdx.x * 128 + wave * 16;

  int r = m0 + l15;
  if (r > NN - 1) r = NN - 1;       // clamp loads; stores predicated (EXEC all-1 at WMMA)
  const float* hr = h   + (size_t)r * DD;
  const float* gr = agg + (size_t)r * DD;

  // ---- A fragments for GEMM1 (K = 128 -> 4 x k32), z = h + agg fused ----
  v16h A1[4];
#pragma unroll
  for (int k = 0; k < 4; ++k) {
    int kb = k * 32;
    A1[k] = gin_join(gin_cvt8(hr + kb + off,      gr + kb + off),
                     gin_cvt8(hr + kb + 16 + off, gr + kb + 16 + off));
  }

  // ---- GEMM1: 16 N-tiles of hidden; bias + LeakyReLU; spill f16 tile to LDS ----
  _Float16* hid = sHid + wave * 16 * HH;
#pragma unroll 2
  for (int n = 0; n < 16; ++n) {
    float bias = b1g[n * 16 + l15];
    v8f acc;
#pragma unroll
    for (int i = 0; i < 8; ++i) acc[i] = bias;
    const _Float16* wb = sW1t + (size_t)(n * 16 + l15) * DD + off;
#pragma unroll
    for (int k = 0; k < 4; ++k) {
      v16h b = gin_join(*(const v8h*)(wb + k * 32), *(const v8h*)(wb + k * 32 + 16));
      acc = __builtin_amdgcn_wmma_f32_16x16x32_f16(false, A1[k], false, b,
                                                   (short)0, acc, false, false);
    }
#pragma unroll
    for (int i = 0; i < 8; ++i) {
      int m = off + i;                       // C layout: M = 8*hihalf + i
      hid[m * HH + n * 16 + l15] = (_Float16)gin_lrelu(acc[i]);
    }
  }
  // same-wave DS ops are in-order; per-wave private tile -> no barrier needed

  // ---- A fragments for GEMM2 from LDS hidden (K = 256 -> 8 x k32) ----
  const _Float16* ar = hid + l15 * HH;
  v16h A2[8];
#pragma unroll
  for (int k = 0; k < 8; ++k) {
    int kb = k * 32;
    v8h lo = *(const v8h*)(ar + kb + off);
    v8h hi = *(const v8h*)(ar + kb + 16 + off);
    A2[k] = gin_join(lo, hi);
  }

  // ---- GEMM2: 8 N-tiles of output; bias (+ outer LeakyReLU); store f32 ----
#pragma unroll 2
  for (int n = 0; n < 8; ++n) {
    float bias = b2g[n * 16 + l15];
    v8f acc;
#pragma unroll
    for (int i = 0; i < 8; ++i) acc[i] = bias;
    const _Float16* wb = sW2t + (size_t)(n * 16 + l15) * HH + off;
#pragma unroll
    for (int k = 0; k < 8; ++k) {
      v16h b = gin_join(*(const v8h*)(wb + k * 32), *(const v8h*)(wb + k * 32 + 16));
      acc = __builtin_amdgcn_wmma_f32_16x16x32_f16(false, A2[k], false, b,
                                                   (short)0, acc, false, false);
    }
#pragma unroll
    for (int i = 0; i < 8; ++i) {
      float v = acc[i];
      if (outer_relu) v = gin_lrelu(v);
      int g = m0 + off + i;
      if (g < NN) out[(size_t)g * DD + n * 16 + l15] = v;
    }
  }

  // ---- fused agg re-zero: each wave zeroes ONLY its own 16 rows (no barrier needed;
  //      this wave's agg reads completed before its first WMMA consumed them) ----
#pragma unroll
  for (int i4 = 0; i4 < 16; ++i4) {          // 16*32 float4 per wave / 32 lanes
    int idx = i4 * 32 + lane;                // 0..511
    int row = m0 + (idx >> 5);               // 32 float4 per row
    if (row < NN)
      ((float4*)(agg + (size_t)row * DD))[idx & 31] = make_float4(0.f, 0.f, 0.f, 0.f);
  }
}

extern "C" void kernel_launch(void* const* d_in, const int* in_sizes, int n_in,
                              void* d_out, int out_size, void* d_ws, size_t ws_size,
                              hipStream_t stream) {
  const float*     x  = (const float*)d_in[0];
  const long long* ei = (const long long*)d_in[1];
  const float*     W1 = (const float*)d_in[2];   // [5][128][256]
  const float*     b1 = (const float*)d_in[3];   // [5][256]
  const float*     W2 = (const float*)d_in[4];   // [5][256][128]
  const float*     b2 = (const float*)d_in[5];   // [5][128]
  float* out = (float*)d_out;

  float* bufA = (float*)d_ws;                    // ping
  float* bufB = bufA + (size_t)NN * DD;          // pong
  float* agg  = bufB + (size_t)NN * DD;          // aggregate buffer

  const size_t lds_bytes =
      (size_t)(HH * DD + DD * HH + 128 * HH) * sizeof(_Float16);   // 192 KB

  const int zero_n4   = NN * DD / 4;
  const int zero_grid = (zero_n4 + 255) / 256;
  const int scat_grid = (NE * 32 + 255) / 256;
  const int mlp_grid  = (NN + 127) / 128;

  // agg must start zeroed (d_ws is poisoned); layers 1..4 get agg re-zeroed by
  // the previous layer's MLP kernel (fused, per-wave-private rows).
  gin_zero_kernel<<<zero_grid, 256, 0, stream>>>(agg, zero_n4);

  const float* hcur = x;
  for (int l = 0; l < 5; ++l) {
    gin_scatter_kernel<<<scat_grid, 256, 0, stream>>>(hcur, ei, agg);
    float* hnext = (l == 4) ? out : ((l & 1) ? bufB : bufA);
    gin_mlp_kernel<<<mlp_grid, 256, lds_bytes, stream>>>(
        hcur, agg, hnext,
        W1 + (size_t)l * DD * HH, b1 + (size_t)l * HH,
        W2 + (size_t)l * HH * DD, b2 + (size_t)l * DD,
        (l < 4) ? 1 : 0);
    hcur = hnext;
  }
}
